// CrossAttentionModule_50603304682023
// MI455X (gfx1250) — matrix operand order repaired and checked by reading, not compiled, not run on hardware
//
#include <hip/hip_runtime.h>
#include <hip/hip_bf16.h>
#include <math.h>

typedef _Float16 f16;
typedef __attribute__((ext_vector_type(16))) _Float16 v16h;
typedef __attribute__((ext_vector_type(8)))  _Float16 v8h;
typedef __attribute__((ext_vector_type(4)))  _Float16 v4h;
typedef __attribute__((ext_vector_type(8)))  float    v8f;
typedef __attribute__((ext_vector_type(4)))  float    v4f;

#define EMBED 512
#define NBATCH 8
#define QLEN 2048
#define KLEN 4096

// ---------------------------------------------------------------------------
// Fragment loaders for C = A * B^T, both row-major, K contiguous.
// A-frag (16x32 f16): lane L holds row M = row0 + (L&15).
//   lanes 0-15 : VGPR0-3 = K 0..7,  VGPR4-7 = K 16..23
//   lanes 16-31: VGPR0-3 = K 8..15, VGPR4-7 = K 24..31
// ---------------------------------------------------------------------------
__device__ __forceinline__ v16h load_a_frag(const f16* __restrict__ A, int lda,
                                            int row0, int k0, int lane) {
    int m = row0 + (lane & 15);
    int g = lane >> 4;
    const f16* p = A + (size_t)m * lda + k0 + g * 8;
    v8h lo = *(const v8h*)(p);
    v8h hi = *(const v8h*)(p + 16);
    return __builtin_shufflevector(lo, hi, 0,1,2,3,4,5,6,7,8,9,10,11,12,13,14,15);
}

// B-frag (32x16 f16, used as B^T): lane L holds column N = col0 + (L&15),
// same K layout as A-frag: lanes 0-15 K 0..7,16..23; lanes 16-31 K 8..15,24..31.
__device__ __forceinline__ v16h load_b_frag(const f16* __restrict__ B, int ldb,
                                            int col0, int k0, int lane) {
    int n = col0 + (lane & 15);
    int g = lane >> 4;
    const f16* p = B + (size_t)n * ldb + k0 + g * 8;
    v8h lo = *(const v8h*)(p);
    v8h hi = *(const v8h*)(p + 16);
    return __builtin_shufflevector(lo, hi, 0,1,2,3,4,5,6,7,8,9,10,11,12,13,14,15);
}

// C-frag store (16x16 f32 acc): VGPR r -> row = row0 + r + 8*(lane>>4), col = col0 + (lane&15)
__device__ __forceinline__ void store_c_f16(f16* __restrict__ C, int ldc,
                                            int row0, int col0, v8f c, int lane) {
    int n  = col0 + (lane & 15);
    int mb = row0 + 8 * (lane >> 4);
#pragma unroll
    for (int r = 0; r < 8; ++r)
        C[(size_t)(mb + r) * ldc + n] = (f16)c[r];
}

#define WMMA_F16(a, b, c) \
    __builtin_amdgcn_wmma_f32_16x16x32_f16(false, (a), false, (b), (short)0, (c), false, false)

// ---------------------------------------------------------------------------
// Batched GEMM: C[b] = A[b] * B[b]^T   (f16 in, f32 acc, f16 out)
// Block = 8 waves (2 M x 4 N), wave tile 64x64 (4x4 wmma frags, 16 accumulators),
// block tile 128(M) x 256(N). One b128 load per WMMA -> ~32 flop/byte from L2.
// Requires M%128==0, N%256==0, K%32==0 (true for all calls here).
// ---------------------------------------------------------------------------
__global__ __launch_bounds__(256)
void gemm_abt_f16(const f16* __restrict__ A, long long sA,
                  const f16* __restrict__ B, long long sB,
                  f16* __restrict__ C, long long sC,
                  int K, int lda, int ldb, int ldc) {
    const int lane = threadIdx.x & 31;
    const int wave = threadIdx.x >> 5;
    const int row0 = blockIdx.y * 128 + (wave >> 2) * 64;
    const int col0 = blockIdx.x * 256 + (wave & 3) * 64;
    const f16* Ab = A + (size_t)blockIdx.z * sA;
    const f16* Bb = B + (size_t)blockIdx.z * sB;
    f16*       Cb = C + (size_t)blockIdx.z * sC;

    v8f acc[4][4] = {};
    for (int k0 = 0; k0 < K; k0 += 32) {
        v16h af[4], bf[4];
#pragma unroll
        for (int i = 0; i < 4; ++i) af[i] = load_a_frag(Ab, lda, row0 + 16 * i, k0, lane);
#pragma unroll
        for (int j = 0; j < 4; ++j) bf[j] = load_b_frag(Bb, ldb, col0 + 16 * j, k0, lane);
#pragma unroll
        for (int i = 0; i < 4; ++i)
#pragma unroll
            for (int j = 0; j < 4; ++j)
                acc[i][j] = WMMA_F16(af[i], bf[j], acc[i][j]);
    }
#pragma unroll
    for (int i = 0; i < 4; ++i)
#pragma unroll
        for (int j = 0; j < 4; ++j)
            store_c_f16(Cb, ldc, row0 + 16 * i, col0 + 16 * j, acc[i][j], lane);
}

// ---------------------------------------------------------------------------
// Final GEMM: out = xres + O * Wo^T + bo   (f32 output, residual + bias epilogue)
// Same 64x64 wave tiling.
// ---------------------------------------------------------------------------
__global__ __launch_bounds__(256)
void gemm_abt_bias_res(const f16* __restrict__ A, const f16* __restrict__ B,
                       const float* __restrict__ xres, const float* __restrict__ bo,
                       float* __restrict__ Out,
                       int K, int lda, int ldb, int ldc) {
    const int lane = threadIdx.x & 31;
    const int wave = threadIdx.x >> 5;
    const int row0 = blockIdx.y * 128 + (wave >> 2) * 64;
    const int col0 = blockIdx.x * 256 + (wave & 3) * 64;

    v8f acc[4][4] = {};
    for (int k0 = 0; k0 < K; k0 += 32) {
        v16h af[4], bf[4];
#pragma unroll
        for (int i = 0; i < 4; ++i) af[i] = load_a_frag(A, lda, row0 + 16 * i, k0, lane);
#pragma unroll
        for (int j = 0; j < 4; ++j) bf[j] = load_b_frag(B, ldb, col0 + 16 * j, k0, lane);
#pragma unroll
        for (int i = 0; i < 4; ++i)
#pragma unroll
            for (int j = 0; j < 4; ++j)
                acc[i][j] = WMMA_F16(af[i], bf[j], acc[i][j]);
    }
    const int g  = lane >> 4;
    const int nl = lane & 15;
#pragma unroll
    for (int i = 0; i < 4; ++i) {
#pragma unroll
        for (int j = 0; j < 4; ++j) {
            int n   = col0 + 16 * j + nl;
            int mb  = row0 + 16 * i + 8 * g;
            float b = bo[n];
#pragma unroll
            for (int r = 0; r < 8; ++r) {
                size_t idx = (size_t)(mb + r) * ldc + n;
                Out[idx] = xres[idx] + acc[i][j][r] + b;
            }
        }
    }
}

// ---------------------------------------------------------------------------
// In-place row softmax over 4096 f16 columns, with 1/sqrt(512) pre-scale.
// One 256-thread block per row, 16 elements per thread.
// ---------------------------------------------------------------------------
__global__ __launch_bounds__(256)
void softmax_rows_4096(f16* __restrict__ S) {
    __shared__ float red[256];
    const int tid = threadIdx.x;
    const float scale = 0.04419417382415922f;  // 1/sqrt(512)
    f16* row = S + (size_t)blockIdx.x * 4096 + tid * 16;

    v8h d0 = *(v8h*)(row);
    v8h d1 = *(v8h*)(row + 8);
    float v[16];
#pragma unroll
    for (int i = 0; i < 8; ++i) { v[i] = (float)d0[i] * scale; v[i + 8] = (float)d1[i] * scale; }

    float mx = -3.0e38f;
#pragma unroll
    for (int i = 0; i < 16; ++i) mx = fmaxf(mx, v[i]);
    red[tid] = mx; __syncthreads();
    for (int s = 128; s > 0; s >>= 1) {
        if (tid < s) red[tid] = fmaxf(red[tid], red[tid + s]);
        __syncthreads();
    }
    mx = red[0]; __syncthreads();

    float sum = 0.f;
#pragma unroll
    for (int i = 0; i < 16; ++i) { v[i] = __expf(v[i] - mx); sum += v[i]; }
    red[tid] = sum; __syncthreads();
    for (int s = 128; s > 0; s >>= 1) {
        if (tid < s) red[tid] += red[tid + s];
        __syncthreads();
    }
    float inv = 1.0f / red[0];

    v8h o0, o1;
#pragma unroll
    for (int i = 0; i < 8; ++i) { o0[i] = (f16)(v[i] * inv); o1[i] = (f16)(v[i + 8] * inv); }
    *(v8h*)(row)     = o0;
    *(v8h*)(row + 8) = o1;
}

// ---------------------------------------------------------------------------
// Vectorized f32 -> f16 conversion: 8 elements per thread per trip.
__global__ __launch_bounds__(256)
void f32_to_f16(const float* __restrict__ src, f16* __restrict__ dst, size_t n8) {
    size_t i = (size_t)blockIdx.x * blockDim.x + threadIdx.x;
    size_t stride = (size_t)gridDim.x * blockDim.x;
    for (; i < n8; i += stride) {
        v4f a = *(const v4f*)(src + i * 8);
        v4f b = *(const v4f*)(src + i * 8 + 4);
        v8h o;
#pragma unroll
        for (int k = 0; k < 4; ++k) { o[k] = (f16)a[k]; o[k + 4] = (f16)b[k]; }
        *(v8h*)(dst + i * 8) = o;
    }
}

// ---------------------------------------------------------------------------
extern "C" void kernel_launch(void* const* d_in, const int* in_sizes, int n_in,
                              void* d_out, int out_size, void* d_ws, size_t ws_size,
                              hipStream_t stream) {
    (void)in_sizes; (void)n_in; (void)out_size; (void)ws_size;
    const float* x  = (const float*)d_in[0];   // [8,2048,512]
    const float* y  = (const float*)d_in[1];   // [8,4096,512]
    const float* Wq = (const float*)d_in[2];   // [512,512]
    const float* Wk = (const float*)d_in[3];
    const float* Wv = (const float*)d_in[4];
    const float* Wo = (const float*)d_in[5];
    const float* bo = (const float*)d_in[6];   // [512]
    float* out = (float*)d_out;                // [8,2048,512]

    const size_t NQ = (size_t)NBATCH * QLEN;           // 16384
    const size_t NK = (size_t)NBATCH * KLEN;           // 32768
    const size_t nX = NQ * EMBED;                      // 8.39M
    const size_t nY = NK * EMBED;                      // 16.78M
    const size_t nW = (size_t)EMBED * EMBED;           // 262144
    const size_t nS = (size_t)NBATCH * QLEN * KLEN;    // 67.1M

    // carve the f16 workspace
    f16* p = (f16*)d_ws;
    f16* xh  = p; p += nX;
    f16* yh  = p; p += nY;
    f16* wqh = p; p += nW;
    f16* wkh = p; p += nW;
    f16* wvh = p; p += nW;
    f16* woh = p; p += nW;
    f16* Qh  = p; p += nX;             // [NQ, 512]
    f16* Kh  = p; p += nY;             // [NK, 512]
    f16* Vt  = p; p += nY;             // per batch [512, 4096] (V transposed)
    f16* S   = p; p += nS;             // per batch [2048, 4096], softmaxed in place
    f16* O   = p; p += nX;             // [NQ, 512]

    // 1) downconvert inputs/weights to f16 (8 elems/thread)
    f32_to_f16<<<2048, 256, 0, stream>>>(x,  xh,  nX / 8);
    f32_to_f16<<<4096, 256, 0, stream>>>(y,  yh,  nY / 8);
    f32_to_f16<<<128,  256, 0, stream>>>(Wq, wqh, nW / 8);
    f32_to_f16<<<128,  256, 0, stream>>>(Wk, wkh, nW / 8);
    f32_to_f16<<<128,  256, 0, stream>>>(Wv, wvh, nW / 8);
    f32_to_f16<<<128,  256, 0, stream>>>(Wo, woh, nW / 8);

    // 2) Q = x @ Wq^T   [16384,512]
    gemm_abt_f16<<<dim3(EMBED / 256, NQ / 128, 1), 256, 0, stream>>>(
        xh, 0, wqh, 0, Qh, 0, EMBED, EMBED, EMBED, EMBED);
    // 3) K = y @ Wk^T   [32768,512]
    gemm_abt_f16<<<dim3(EMBED / 256, NK / 128, 1), 256, 0, stream>>>(
        yh, 0, wkh, 0, Kh, 0, EMBED, EMBED, EMBED, EMBED);
    // 4) Vt[b] = Wv @ y[b]^T   per batch [512,4096]  (gives V transposed)
    gemm_abt_f16<<<dim3(KLEN / 256, EMBED / 128, NBATCH), 256, 0, stream>>>(
        wvh, 0, yh, (long long)KLEN * EMBED, Vt, (long long)EMBED * KLEN,
        EMBED, EMBED, EMBED, KLEN);
    // 5) S[b] = Q[b] @ K[b]^T   per batch [2048,4096]
    gemm_abt_f16<<<dim3(KLEN / 256, QLEN / 128, NBATCH), 256, 0, stream>>>(
        Qh, (long long)QLEN * EMBED, Kh, (long long)KLEN * EMBED,
        S, (long long)QLEN * KLEN, EMBED, EMBED, EMBED, KLEN);
    // 6) P = softmax(S / sqrt(512)) over keys, in place (16384 rows)
    softmax_rows_4096<<<(unsigned)(NBATCH * QLEN), 256, 0, stream>>>(S);
    // 7) O[b] = P[b] @ Vt[b]^T   per batch [2048,512], contraction over 4096 keys
    gemm_abt_f16<<<dim3(EMBED / 256, QLEN / 128, NBATCH), 256, 0, stream>>>(
        S, (long long)QLEN * KLEN, Vt, (long long)EMBED * KLEN,
        O, (long long)QLEN * EMBED, KLEN, KLEN, KLEN, EMBED);
    // 8) out = x + O @ Wo^T + bo   [16384,512] f32
    gemm_abt_bias_res<<<dim3(EMBED / 256, NQ / 128, 1), 256, 0, stream>>>(
        O, woh, x, bo, out, EMBED, EMBED, EMBED, EMBED);
}
